// cConv_46299747450935
// MI455X (gfx1250) — matrix operand order repaired and checked by reading, not compiled, not run on hardware
//
#include <hip/hip_runtime.h>

typedef __attribute__((ext_vector_type(16))) __bf16 v16bf;
typedef __attribute__((ext_vector_type(8)))  float  v8f;

#define CIN     256
#define COUT    256
#define HH      64
#define WW      64
#define BATCH   32
#define NCC     8                        // ci chunks of 32 (K = tap*256 + ci)
#define PCW     66                       // tile col width: [0]/[65] = zero halo
#define PST     40                       // plane stride (ushorts) per column:
                                         //  20 dwords -> b128 reads hit all 64
                                         //  banks once per lane half
#define WPACK_N (72 * 16 * 32 * 8)       // dwords per weight plane (hi / lo)

__device__ __forceinline__ unsigned short f2bf(float f) {
    unsigned int u = __float_as_uint(f);
    unsigned int r = u + 0x7FFFu + ((u >> 16) & 1u);   // round-to-nearest-even
    return (unsigned short)(r >> 16);
}
__device__ __forceinline__ float bf2f(unsigned short b) {
    return __uint_as_float(((unsigned int)b) << 16);
}

// Pre-pack shared weights into a contiguous-K A-fragment layout (16-bit A,
// 16x32 MxK): VGPR v holds K = base16+2v..2v+1,
// base16 = (lane<16 ? 0 : 16).  hi/lo bf16 planes for
// split-precision fp32 emulation (a_hi*b_hi + a_hi*b_lo + a_lo*b_hi).
__global__ void cconv_prepack_w(const float* __restrict__ wt,
                                unsigned int* __restrict__ wpack) {
    int id = blockIdx.x * blockDim.x + threadIdx.x;     // one packed dword
    if (id >= WPACK_N) return;
    int v    = id & 7;
    int lane = (id >> 3) & 31;
    int t    = (id >> 8) & 15;          // co tile (16 co each)
    int kc   = id >> 12;                // k-chunk 0..71  (kc = tap*8 + ci_chunk)
    int base = (lane & 16) ? 16 : 0;
    int kl0  = base + 2 * v;
    int tap  = kc >> 3;                 // kh*3+kw
    int ci0  = (kc & 7) * 32;
    int co   = t * 16 + (lane & 15);
    const float* wp = wt + (long)co * CIN * 9 + tap;    // [1,COUT,CIN,3,3]
    float f0 = wp[(long)(ci0 + kl0) * 9];
    float f1 = wp[(long)(ci0 + kl0 + 1) * 9];
    unsigned short h0 = f2bf(f0), h1 = f2bf(f1);
    unsigned short l0 = f2bf(f0 - bf2f(h0)), l1 = f2bf(f1 - bf2f(h1));
    wpack[id]           = (unsigned int)h0 | ((unsigned int)h1 << 16);
    wpack[id + WPACK_N] = (unsigned int)l0 | ((unsigned int)l1 << 16);
}

// Workgroup = 8 waves = 2 co-tiles x 4 pixel-tiles -> 32 co x 64 pixels
// (one full image row h: tap row-bounds uniform, column halo spans the tile).
__global__ void __launch_bounds__(256, 2) cconv_main(
    const float* __restrict__ x, const unsigned int* __restrict__ wpack,
    const float* __restrict__ bias, const float* __restrict__ scales,
    const float* __restrict__ shifts, const int* __restrict__ cls,
    float* __restrict__ out)
{
    // f32 async-staging buffers: half ci-chunk (16 ci) x 3 rows x 64 cols, x2
    __shared__ __align__(16) float f32buf[2][3][16][64];            // 24 KB
    // once-converted bf16 planes, TRANSPOSED (ci innermost) so a B fragment
    // is 16 consecutive ushorts -> 2x ds_load_b128, no v_perm, no hazards
    __shared__ __align__(16) unsigned short hiP[3][PCW][PST];       // ~15.5 KB
    __shared__ __align__(16) unsigned short loP[3][PCW][PST];       // ~15.5 KB
    __shared__ float shift_lds[CIN];
    __shared__ float T[PCW];     // per-column  sum_ci shift[ci]*x  accumulator

    const int tid    = threadIdx.x;
    const int lane   = tid & 31;
    const int wave   = tid >> 5;
    const int g      = wave & 3;      // pixel group (16 cols each)
    const int co_idx = wave >> 2;     // 0..1

    const int bid     = blockIdx.x;
    const int h       = bid & 63;
    const int co_grp  = (bid >> 6) & 7;
    const int b       = bid >> 9;
    const int c       = cls[b];
    const int co_tile = co_grp * 2 + co_idx;   // 0..15

    shift_lds[tid] = shifts[(long)c * CIN + tid];
    if (tid < PCW) T[tid] = 0.f;
    if (tid < 192) {                 // zero halo columns of both planes, once
        int dw = tid & 15, side = (tid >> 4) & 1, rp = tid >> 5;    // 0..5
        int rl = rp < 3 ? rp : rp - 3;
        unsigned short* pl = (rp < 3) ? &hiP[rl][side ? PCW - 1 : 0][0]
                                      : &loP[rl][side ? PCW - 1 : 0][0];
        *(unsigned int*)&pl[dw * 2] = 0u;
    }

    // uniform clamped rows for the 3 vertical taps
    int rowv[3], rok[3];
    #pragma unroll
    for (int r = 0; r < 3; ++r) {
        int rr = h + r - 1;
        rok[r]  = (unsigned)rr < (unsigned)HH;
        rowv[r] = rr < 0 ? 0 : (rr >= HH ? HH - 1 : rr);
    }

    const unsigned long long xbase =
        (unsigned long long)(x + (long)b * CIN * HH * WW);

    // Fire-and-forget async stage of one 16-ci half-chunk into LDS (raw f32).
    // 768 b128 transfers / 256 threads = 3 per thread; tracked by ASYNCcnt.
    auto stage = [&](int hc, int buf) {
        const int cibase = hc * 16;
        #pragma unroll
        for (int j = 0; j < 3; ++j) {
            int idx = tid + 256 * j;          // [rl(3)][ci(16)][xv(16)]
            int xv  = idx & 15;
            int ci  = (idx >> 4) & 15;
            int rl  = idx >> 8;
            unsigned goff  = (unsigned)((((cibase + ci) * HH + rowv[rl]) * WW
                                         + xv * 4) * 4);
            unsigned laddr = (unsigned)(size_t)&f32buf[buf][rl][ci][xv * 4];
            asm volatile("global_load_async_to_lds_b128 %0, %1, %2"
                         :: "v"(laddr), "v"(goff), "s"(xbase) : "memory");
        }
    };

    v8f acc_hh = {};   // a_hi*b_hi
    v8f acc_hl = {};   // a_hi*b_lo
    v8f acc_lh = {};   // a_lo*b_hi

    const int n    = lane & 15;
    const int half = lane >> 4;
    const int gcol = g * 16 + n;       // this lane's output column

    stage(0, 0);

    for (int hc = 0; hc < 2 * NCC; ++hc) {
        asm volatile("s_wait_asynccnt 0x0" ::: "memory");
        __syncthreads();                       // staged f32 visible to all
        if (hc + 1 < 2 * NCC) stage(hc + 1, (hc + 1) & 1);

        // ---- convert once: f32 -> transposed bf16 hi/lo planes + T atomics.
        // Threads own (ci-pair x col-pair): each packed dword (two adjacent-ci
        // bf16) is a single b32 store; one atomicAdd per ci-pair per column.
        {
            const int buf  = hc & 1;
            const int hsel = (hc & 1) * 16;    // which ci half of the planes
            #pragma unroll
            for (int j = 0; j < 3; ++j) {
                int idx  = tid + 256 * j;      // [rl(3)][cp(8)][colp(32)]
                int colp = idx & 31;
                int cp   = (idx >> 5) & 7;
                int rl   = idx >> 8;
                const int ci_l  = cp * 2;
                const int col0  = colp * 2;
                const int cidst = hsel + ci_l;
                if (rok[rl]) {
                    const float2 fa = *(const float2*)&f32buf[buf][rl][ci_l][col0];
                    const float2 fb = *(const float2*)&f32buf[buf][rl][ci_l + 1][col0];
                    const float sv0 = shift_lds[hc * 16 + ci_l];
                    const float sv1 = shift_lds[hc * 16 + ci_l + 1];
                    #pragma unroll
                    for (int e = 0; e < 2; ++e) {
                        const float f0 = e ? fa.y : fa.x;
                        const float f1 = e ? fb.y : fb.x;
                        unsigned short h0 = f2bf(f0), h1 = f2bf(f1);
                        unsigned short l0 = f2bf(f0 - bf2f(h0));
                        unsigned short l1 = f2bf(f1 - bf2f(h1));
                        const int pcol = 1 + col0 + e;
                        *(unsigned int*)&hiP[rl][pcol][cidst] =
                            (unsigned)h0 | ((unsigned)h1 << 16);
                        *(unsigned int*)&loP[rl][pcol][cidst] =
                            (unsigned)l0 | ((unsigned)l1 << 16);
                        atomicAdd(&T[pcol], sv0 * f0 + sv1 * f1);
                    }
                } else {                       // zero-pad out-of-image rows
                    #pragma unroll
                    for (int e = 0; e < 2; ++e) {
                        const int pcol = 1 + col0 + e;
                        *(unsigned int*)&hiP[rl][pcol][cidst] = 0u;
                        *(unsigned int*)&loP[rl][pcol][cidst] = 0u;
                    }
                }
            }
        }

        if ((hc & 1) == 0) continue;           // need both halves before GEMM
        __syncthreads();                       // packed planes complete
        const int cc = hc >> 1;

        // ---- 9 taps x 3 split-WMMAs over this 32-ci chunk
        #pragma unroll
        for (int tap = 0; tap < 9; ++tap) {
            const int kh = tap / 3, kw = tap - kh * 3;
            const int pcol = gcol + kw;        // halo makes this always valid

            // B fragment (16-bit B, 32x16 KxN): lane = column n, K = half*16+i
            // -> 16 consecutive ushorts per plane: 2x ds_load_b128 each.
            union { uint4 q[2]; v16bf v; } Bh, Bl;
            const uint4* bhp = (const uint4*)&hiP[kh][pcol][half * 16];
            const uint4* blp = (const uint4*)&loP[kh][pcol][half * 16];
            Bh.q[0] = bhp[0]; Bh.q[1] = bhp[1];
            Bl.q[0] = blp[0]; Bl.q[1] = blp[1];

            // A fragments: pre-packed, 2x b128 per plane
            union { uint4 q[2]; v16bf v; } Ah, Al;
            const int kc = tap * 8 + cc;
            const uint4* ap  = (const uint4*)(wpack +
                (((long)kc * 16 + co_tile) * 32 + lane) * 8);
            const uint4* alp = (const uint4*)((const unsigned int*)ap + WPACK_N);
            Ah.q[0] = ap[0];  Ah.q[1] = ap[1];
            Al.q[0] = alp[0]; Al.q[1] = alp[1];

            acc_hh = __builtin_amdgcn_wmma_f32_16x16x32_bf16(
                false, Ah.v, false, Bh.v, (short)0, acc_hh, false, false);
            acc_hl = __builtin_amdgcn_wmma_f32_16x16x32_bf16(
                false, Ah.v, false, Bl.v, (short)0, acc_hl, false, false);
            acc_lh = __builtin_amdgcn_wmma_f32_16x16x32_bf16(
                false, Al.v, false, Bh.v, (short)0, acc_lh, false, false);
        }
    }

    // shift field: 3x3 box-sum separates -> S[p] = T[p-1]+T[p]+T[p+1]
    // (T index is halo-shifted by +1; all atomics completed before the last
    // pre-GEMM barrier)
    const float s_pix = T[gcol] + T[gcol + 1] + T[gcol + 2];

    // Epilogue per C/D layout: VGPR r -> row r (lanes 0-15) or r+8 (16-31)
    float* op = out + (((long)b * COUT) * HH + h) * WW + gcol;
    const float* scrow = scales + (long)c * COUT;
    #pragma unroll
    for (int r = 0; r < 8; ++r) {
        const int co_r = co_tile * 16 + r + half * 8;
        const float conv = acc_hh[r] + acc_hl[r] + acc_lh[r];
        op[(long)co_r * (HH * WW)] = conv * scrow[co_r] + s_pix + bias[co_r];
    }
}

extern "C" void kernel_launch(void* const* d_in, const int* in_sizes, int n_in,
                              void* d_out, int out_size, void* d_ws, size_t ws_size,
                              hipStream_t stream) {
    const float* x      = (const float*)d_in[0];
    const float* wt     = (const float*)d_in[1];
    const float* bias   = (const float*)d_in[2];
    const float* scales = (const float*)d_in[3];
    const float* shifts = (const float*)d_in[4];
    const int*   cls    = (const int*)d_in[5];
    float* out = (float*)d_out;
    unsigned int* wpack = (unsigned int*)d_ws;   // 2 planes * 1.18 MB

    cconv_prepack_w<<<(WPACK_N + 255) / 256, 256, 0, stream>>>(wt, wpack);
    cconv_main<<<BATCH * 8 * 64, 256, 0, stream>>>(
        x, wpack, bias, scales, shifts, cls, out);
}
